// MultipleCRF_48284022342096
// MI455X (gfx1250) — compile-verified, matrix operand order checked by reading, not run on hardware
//
#include <hip/hip_runtime.h>

#define S        64          // N_LABELS * DUPLICITY
#define SEQ      2048
#define NBATCH   256
#define PEN      10000.0f    // reference INF
#define LN2      0.69314718056f

typedef __attribute__((ext_vector_type(16))) _Float16 v16h;
typedef __attribute__((ext_vector_type(8)))  float    v8f;

// ---------------------------------------------------------------------------
// CDNA5 async LDS copy: one instruction moves 16B per lane (512B/wave).
// IOFFSET is added to BOTH the LDS and the global address (ISA 08 §4.4),
// so 8 issues with offset 0..112 stream a 4KB tile (16 rows x 256B).
// ---------------------------------------------------------------------------
__device__ __forceinline__ void async_tile_load(uint32_t lds, const float* g) {
    asm volatile("global_load_async_to_lds_b128 %0, %1, off"            :: "v"(lds), "v"(g) : "memory");
    asm volatile("global_load_async_to_lds_b128 %0, %1, off offset:16"  :: "v"(lds), "v"(g) : "memory");
    asm volatile("global_load_async_to_lds_b128 %0, %1, off offset:32"  :: "v"(lds), "v"(g) : "memory");
    asm volatile("global_load_async_to_lds_b128 %0, %1, off offset:48"  :: "v"(lds), "v"(g) : "memory");
    asm volatile("global_load_async_to_lds_b128 %0, %1, off offset:64"  :: "v"(lds), "v"(g) : "memory");
    asm volatile("global_load_async_to_lds_b128 %0, %1, off offset:80"  :: "v"(lds), "v"(g) : "memory");
    asm volatile("global_load_async_to_lds_b128 %0, %1, off offset:96"  :: "v"(lds), "v"(g) : "memory");
    asm volatile("global_load_async_to_lds_b128 %0, %1, off offset:112" :: "v"(lds), "v"(g) : "memory");
}
// Async loads complete in order: <=8 outstanding means the previous tile landed.
__device__ __forceinline__ void wait_async_le8() {
    asm volatile("s_wait_asynccnt 8" ::: "memory");
}

// ---------------------------------------------------------------------------
// Precompute column maxima of transitions and exp(T - colmax) in f16.
// exp(T - colmax) in (0,1] -> exactly the dynamic range f16 handles well.
// ---------------------------------------------------------------------------
__global__ void crf_prep(const float* __restrict__ trans,
                         float* __restrict__ colmax,
                         _Float16* __restrict__ expT) {
    int j = threadIdx.x;                  // 64 threads, one per column
    float m = -3.4e38f;
    for (int i = 0; i < S; ++i) m = fmaxf(m, trans[i * S + j]);
    colmax[j] = m;
    for (int i = 0; i < S; ++i)
        expT[i * S + j] = (_Float16)__expf(trans[i * S + j] - m);
}

__global__ void crf_zero(float* out) { if (threadIdx.x == 0) out[0] = 0.0f; }

// ---------------------------------------------------------------------------
// One wave32 per (16-batch tile, pass). pass==1 applies the label mask.
// Per step: carry' = log( exp(carry-ma) @ exp(T-mb) ) + ma + mb + x_t
// exp-space matmul = 4 x (2 chained v_wmma_f32_16x16x32_f16);
// x_t tiles double-buffered into LDS via global_load_async_to_lds_b128.
// ---------------------------------------------------------------------------
__global__ __launch_bounds__(32)
void crf_scan(const float* __restrict__ x, const int* __restrict__ y,
              const float* __restrict__ colmax, const _Float16* __restrict__ expT,
              float* __restrict__ out) {
    __shared__ float Xs[2][16][S];     // double-buffered emissions tile
    __shared__ float C[16][S + 8];     // carry, padded rows
    __shared__ float ma_s[16];
    __shared__ float pm[32];
    __shared__ int   ys_s[16];

    const int lane = threadIdx.x;
    const int nn   = lane & 15;        // row (A) / col (B,C) index within tile
    const int hi   = lane >> 4;        // which lane-half
    const int pass = blockIdx.x & 1;   // 0 = free energy, 1 = masked score
    const int b0   = (blockIdx.x >> 1) * 16;

    // Per-lane async-copy bases: lane -> (row m = lane>>1, 128B chunk = lane&1)
    const int am = lane >> 1, ac = lane & 1;
    const float* gbase = x + ((size_t)(b0 + am) * SEQ) * S + ac * 32;
    const uint32_t l0 = (uint32_t)(uintptr_t)(&Xs[0][am][ac * 32]);
    const uint32_t l1 = (uint32_t)(uintptr_t)(&Xs[1][am][ac * 32]);

    // --- B fragments: exp(T - mb), resident for the whole scan -------------
    // 16-bit B layout (ISA 7.12.2): VGPR v, lanes 0-15 hold K in {2v,2v+1}
    // (+16 for v>=4), lanes 16-31 the +8 K group.  kh selects K-half (32).
    v16h Bf[4][2];
#pragma unroll
    for (int nt = 0; nt < 4; ++nt)
#pragma unroll
        for (int kh = 0; kh < 2; ++kh)
#pragma unroll
            for (int h = 0; h < 16; ++h) {
                int v  = h >> 1, odd = h & 1;
                int kl = (v < 4) ? (hi * 8 + 2 * v + odd)
                                 : (16 + hi * 8 + 2 * (v - 4) + odd);
                Bf[nt][kh][h] = expT[(size_t)(kh * 32 + kl) * S + nt * 16 + nn];
            }

    float mbv[4];
#pragma unroll
    for (int nt = 0; nt < 4; ++nt) mbv[nt] = colmax[nt * 16 + nn];

    // --- init carry with x[:,0,:] (+ mask) ---------------------------------
    for (int e = lane; e < 16 * S; e += 32) {
        int m = e >> 6, s = e & 63;
        float v = x[((size_t)(b0 + m) * SEQ + 0) * S + s];
        if (pass && (y[(size_t)(b0 + m) * SEQ + 0] != (s >> 1))) v -= PEN;
        C[m][s] = v;
    }
    // start streaming x[:,1,:] into buffer 1
    async_tile_load(l1, gbase + (size_t)1 * S);
    __syncthreads();

    for (int t = 1; t < SEQ; ++t) {
        // kick off the next tile (clamped re-load on the last step keeps the
        // outstanding-count invariant for s_wait_asynccnt 8)
        const int tn = (t + 1 < SEQ) ? (t + 1) : t;
        async_tile_load(((t + 1) & 1) ? l1 : l0, gbase + (size_t)tn * S);

        // this step's labels (masked pass only; uniform branch)
        if (pass && lane < 16) ys_s[lane] = y[(size_t)(b0 + lane) * SEQ + t];

        // row maxima ma[m] over 64 states (each lane scans half a row)
        float p = -3.4e38f;
#pragma unroll 8
        for (int s2 = 0; s2 < 32; ++s2) p = fmaxf(p, C[nn][hi * 32 + s2]);
        pm[lane] = p;
        __syncthreads();
        if (lane < 16) ma_s[lane] = fmaxf(pm[lane], pm[lane + 16]);
        __syncthreads();

        // A fragments: exp(carry - ma) in f16 (A layout mirrors B's K map)
        const float mam = ma_s[nn];
        v16h Af[2];
#pragma unroll
        for (int kh = 0; kh < 2; ++kh)
#pragma unroll
            for (int h = 0; h < 16; ++h) {
                int v  = h >> 1, odd = h & 1;
                int kl = (v < 4) ? (hi * 8 + 2 * v + odd)
                                 : (16 + hi * 8 + 2 * (v - 4) + odd);
                Af[kh][h] = (_Float16)__expf(C[nn][kh * 32 + kl] - mam);
            }

        // exp-space matmul: [16x64] @ [64x64] -> 4 tiles of 16x16 (f32 acc)
        v8f acc[4];
#pragma unroll
        for (int nt = 0; nt < 4; ++nt) {
            v8f a = {};
            a = __builtin_amdgcn_wmma_f32_16x16x32_f16(false, Af[0], false, Bf[nt][0],
                                                       (short)0, a, false, false);
            a = __builtin_amdgcn_wmma_f32_16x16x32_f16(false, Af[1], false, Bf[nt][1],
                                                       (short)0, a, false, false);
            acc[nt] = a;
        }

        wait_async_le8();   // tile for step t has landed in Xs[t&1]
        __syncthreads();    // all lanes finished reading C

        // carry' = log(acc) + ma + mb + x_t (+ mask)
#pragma unroll
        for (int nt = 0; nt < 4; ++nt) {
#pragma unroll
            for (int r = 0; r < 8; ++r) {
                int M = r + hi * 8;             // C/D layout: vgpr r -> row
                int N = nt * 16 + nn;
                float xv = Xs[t & 1][M][N];
                if (pass && (ys_s[M] != (N >> 1))) xv -= PEN;
                C[M][N] = LN2 * __log2f(fmaxf(acc[nt][r], 1e-37f))
                        + ma_s[M] + mbv[nt] + xv;
            }
        }
        __syncthreads();
    }

    // final logsumexp per batch row -> Q, accumulate +-Q/256 into out
    float p = -3.4e38f;
#pragma unroll 8
    for (int s2 = 0; s2 < 32; ++s2) p = fmaxf(p, C[nn][hi * 32 + s2]);
    pm[lane] = p;
    __syncthreads();
    if (lane < 16) ma_s[lane] = fmaxf(pm[lane], pm[lane + 16]);
    __syncthreads();
    float sum = 0.0f;
#pragma unroll 8
    for (int s2 = 0; s2 < 32; ++s2) sum += __expf(C[nn][hi * 32 + s2] - ma_s[nn]);
    pm[lane] = sum;
    __syncthreads();
    if (lane < 16) {
        float Q = LN2 * __log2f(pm[lane] + pm[lane + 16]) + ma_s[lane];
        float contrib = (pass ? -Q : Q) * (1.0f / (float)NBATCH);
        atomicAdd(out, contrib);
    }
}

// ---------------------------------------------------------------------------
extern "C" void kernel_launch(void* const* d_in, const int* in_sizes, int n_in,
                              void* d_out, int out_size, void* d_ws, size_t ws_size,
                              hipStream_t stream) {
    const float*    x     = (const float*)d_in[0];
    const int*      y     = (const int*)d_in[1];
    const float*    trans = (const float*)d_in[2];
    float*          out   = (float*)d_out;

    float*    colmax = (float*)d_ws;                                   // 64 f32
    _Float16* expT   = (_Float16*)((char*)d_ws + 64 * sizeof(float));  // 64x64 f16

    crf_zero<<<1, 32, 0, stream>>>(out);
    crf_prep<<<1, 64, 0, stream>>>(trans, colmax, expT);
    // 16 batch tiles x 2 passes; one wave32 per block (WMMA is per-wave)
    crf_scan<<<32, 32, 0, stream>>>(x, y, colmax, expT, out);
}
